// CausalSelfAttention_18296560681743
// MI455X (gfx1250) — matrix operand order repaired and checked by reading, not compiled, not run on hardware
//
#include <hip/hip_runtime.h>
#include <hip/hip_bf16.h>
#include <stddef.h>

typedef __attribute__((ext_vector_type(16))) __bf16 bf16x16;
typedef __attribute__((ext_vector_type(8)))  __bf16 bf16x8;
typedef __attribute__((ext_vector_type(8)))  float  f32x8;

#define N_EMBD 768
#define N_HEAD 12
#define HEAD_DIM 64
#define TSEQ 4096
#define BATCH 2
#define BT (BATCH * TSEQ)      // 8192 rows
#define C3 (3 * N_EMBD)        // 2304

// ---------------- prep kernels ----------------

__global__ void cvt_f32_bf16(const float* __restrict__ in, __bf16* __restrict__ out, int n) {
    int i = blockIdx.x * blockDim.x + threadIdx.x;
    if (i < n) out[i] = (__bf16)in[i];
}

// in: [K][N] f32 row-major  ->  out: [N][K] bf16 row-major
__global__ void transpose_cvt(const float* __restrict__ in, __bf16* __restrict__ out,
                              int K, int N) {
    int i = blockIdx.x * blockDim.x + threadIdx.x;
    if (i >= K * N) return;
    int k = i / N, n = i % N;
    out[(size_t)n * K + k] = (__bf16)in[i];
}

// helper: build 16-bit A fragment from two contiguous 8-element halves
__device__ __forceinline__ bf16x16 make_afrag(const __bf16* p) {
    bf16x8 lo = *(const bf16x8*)p;
    bf16x8 hi = *(const bf16x8*)(p + 8);
    bf16x16 a;
    for (int i = 0; i < 8; ++i) { a[i] = lo[i]; a[8 + i] = hi[i]; }
    return a;
}

// ---------------- QKV GEMM: xb[8192x768] @ W_attn -> Q,K (B,H,T,D) and Vt (B,H,D,T), bf16 ----------------
// 32x64 output tile per wave: 2 A-frags x 4 B-frags, B loads shared across A rows.

__global__ __launch_bounds__(32)
void qkv_gemm(const __bf16* __restrict__ xb, const __bf16* __restrict__ waT,
              __bf16* __restrict__ Q, __bf16* __restrict__ Kk, __bf16* __restrict__ Vt) {
    const int lane = threadIdx.x & 31;
    const int half = lane >> 4;
    const int l15  = lane & 15;
    const int row0 = blockIdx.x * 32;           // 256 M tiles of 32 rows
    const int nb   = blockIdx.y;                // 36 blocks of 64 cols

    f32x8 c[2][4];
    for (int m = 0; m < 2; ++m)
        for (int j = 0; j < 4; ++j)
            for (int r = 0; r < 8; ++r) c[m][j][r] = 0.0f;

    const __bf16* arow0 = xb + (size_t)(row0 + l15) * N_EMBD;
    const __bf16* arow1 = arow0 + (size_t)16 * N_EMBD;
    for (int k0 = 0; k0 < N_EMBD; k0 += 32) {
        // A fragments: m = l15, k = i + 16*half
        bf16x16 a0 = make_afrag(arow0 + k0 + half * 16);
        bf16x16 a1 = make_afrag(arow1 + k0 + half * 16);
        for (int j = 0; j < 4; ++j) {
            int n = nb * 64 + j * 16 + l15;     // B fragment: n = l15, k = i + 16*half
            const __bf16* bp = waT + (size_t)n * N_EMBD + k0 + half * 16;
            bf16x16 b = *(const bf16x16*)bp;
            c[0][j] = __builtin_amdgcn_wmma_f32_16x16x32_bf16(false, a0, false, b,
                                                              (short)0, c[0][j], false, false);
            c[1][j] = __builtin_amdgcn_wmma_f32_16x16x32_bf16(false, a1, false, b,
                                                              (short)0, c[1][j], false, false);
        }
    }
    // scatter into head-major layouts
    for (int m = 0; m < 2; ++m) {
        for (int j = 0; j < 4; ++j) {
            int n   = nb * 64 + j * 16 + l15;
            int sel = n / N_EMBD;               // 0=q 1=k 2=v
            int cc  = n % N_EMBD;
            int h = cc / HEAD_DIM, d = cc % HEAD_DIM;
            for (int r = 0; r < 8; ++r) {
                int row = row0 + m * 16 + r + half * 8;
                int bq  = row >> 12;            // /4096
                int t   = row & 4095;
                size_t bh = (size_t)(bq * N_HEAD + h);
                __bf16 val = (__bf16)c[m][j][r];
                if (sel == 0)      Q [(bh * TSEQ + t) * HEAD_DIM + d] = val;
                else if (sel == 1) Kk[(bh * TSEQ + t) * HEAD_DIM + d] = val;
                else               Vt[(bh * HEAD_DIM + d) * TSEQ + t] = val;
            }
        }
    }
}

// ---------------- flash attention: one wave per 32 query rows per (b,h) ----------------
// K/V fragments loaded once per 32-key block and reused by both query row-tiles.

__global__ __launch_bounds__(32)
void flash_attn(const __bf16* __restrict__ Q, const __bf16* __restrict__ Kk,
                const __bf16* __restrict__ Vt, __bf16* __restrict__ yb) {
    __shared__ __bf16 P[2][16][32];

    const int lane = threadIdx.x & 31;
    const int half = lane >> 4;
    const int l15  = lane & 15;
    const int q0   = blockIdx.x * 32;           // 128 query blocks of 32 rows
    const int bh   = blockIdx.y;                // 24 (b,h) pairs
    const int b    = bh / N_HEAD;
    const int h    = bh % N_HEAD;

    const __bf16* Qh = Q  + (size_t)bh * TSEQ * HEAD_DIM;
    const __bf16* Kh = Kk + (size_t)bh * TSEQ * HEAD_DIM;
    const __bf16* Vh = Vt + (size_t)bh * HEAD_DIM * TSEQ;

    // Q A-fragments: [qtile][d-halves]
    bf16x16 aq[2][2];
    for (int qt = 0; qt < 2; ++qt) {
        const __bf16* qrow = Qh + (size_t)(q0 + qt * 16 + l15) * HEAD_DIM;
        for (int kk = 0; kk < 2; ++kk)
            aq[qt][kk] = make_afrag(qrow + kk * 32 + half * 16);
    }

    float mrow[2][8], lrow[2][8];
    f32x8 o[2][4];
    for (int qt = 0; qt < 2; ++qt)
        for (int r = 0; r < 8; ++r) { mrow[qt][r] = -3.0e38f; lrow[qt][r] = 0.0f; }
    for (int qt = 0; qt < 2; ++qt)
        for (int j = 0; j < 4; ++j)
            for (int r = 0; r < 8; ++r) o[qt][j][r] = 0.0f;

    const float scale = 0.125f;                 // 1/sqrt(64)
    const int nkb = q0 / 32 + 1;                // ceil((q0+32)/32)

    for (int kb = 0; kb < nkb; ++kb) {
        const int j0 = kb * 32;
        // S = Q K^T : K fragments loaded once, reused by both q-tiles
        f32x8 s[2][2];
        for (int tt = 0; tt < 2; ++tt) {
            int key = j0 + tt * 16 + l15;
            const __bf16* kr = Kh + (size_t)key * HEAD_DIM;
            bf16x16 bk0 = *(const bf16x16*)(kr + half * 16);
            bf16x16 bk1 = *(const bf16x16*)(kr + 32 + half * 16);
            for (int qt = 0; qt < 2; ++qt) {
                f32x8 acc;
                for (int r = 0; r < 8; ++r) acc[r] = 0.0f;
                acc = __builtin_amdgcn_wmma_f32_16x16x32_bf16(false, aq[qt][0], false, bk0,
                                                              (short)0, acc, false, false);
                acc = __builtin_amdgcn_wmma_f32_16x16x32_bf16(false, aq[qt][1], false, bk1,
                                                              (short)0, acc, false, false);
                s[qt][tt] = acc;
            }
        }
        // online softmax per q-tile; stage P into LDS in A-fragment layout
        __syncthreads();
        for (int qt = 0; qt < 2; ++qt) {
            for (int r = 0; r < 8; ++r) {
                int m  = r + half * 8;
                int tq = q0 + qt * 16 + m;
                float s0 = s[qt][0][r] * scale; if (j0 + l15      > tq) s0 = -3.0e38f;
                float s1 = s[qt][1][r] * scale; if (j0 + 16 + l15 > tq) s1 = -3.0e38f;
                float mb = fmaxf(s0, s1);
                for (int off = 1; off < 16; off <<= 1)
                    mb = fmaxf(mb, __shfl_xor(mb, off, 32));
                float mnew  = fmaxf(mrow[qt][r], mb);
                float alpha = __expf(mrow[qt][r] - mnew);
                float p0 = __expf(s0 - mnew);
                float p1 = __expf(s1 - mnew);
                float rs = p0 + p1;
                for (int off = 1; off < 16; off <<= 1)
                    rs += __shfl_xor(rs, off, 32);
                lrow[qt][r] = lrow[qt][r] * alpha + rs;
                mrow[qt][r] = mnew;
                for (int j = 0; j < 4; ++j) o[qt][j][r] *= alpha;
                P[qt][m][l15]      = (__bf16)p0;
                P[qt][m][16 + l15] = (__bf16)p1;
            }
        }
        __syncthreads();
        bf16x16 apv[2];
        for (int qt = 0; qt < 2; ++qt)
            apv[qt] = make_afrag(&P[qt][l15][half * 16]);
        // O += P V : V fragments loaded once, reused by both q-tiles
        for (int j = 0; j < 4; ++j) {
            int d = j * 16 + l15;
            const __bf16* vr = Vh + (size_t)d * TSEQ + j0 + half * 16;
            bf16x16 bv = *(const bf16x16*)vr;
            o[0][j] = __builtin_amdgcn_wmma_f32_16x16x32_bf16(false, apv[0], false, bv,
                                                              (short)0, o[0][j], false, false);
            o[1][j] = __builtin_amdgcn_wmma_f32_16x16x32_bf16(false, apv[1], false, bv,
                                                              (short)0, o[1][j], false, false);
        }
        __syncthreads();
    }

    // normalize and write y (row-major [8192 x 768], bf16)
    for (int qt = 0; qt < 2; ++qt) {
        for (int j = 0; j < 4; ++j) {
            int col = h * HEAD_DIM + j * 16 + l15;
            for (int r = 0; r < 8; ++r) {
                int t = q0 + qt * 16 + r + half * 8;
                float val = o[qt][j][r] / lrow[qt][r];
                yb[((size_t)(b * TSEQ + t)) * N_EMBD + col] = (__bf16)val;
            }
        }
    }
}

// ---------------- output projection: yb[8192x768] @ W_proj -> f32 out ----------------

__global__ __launch_bounds__(32)
void proj_gemm(const __bf16* __restrict__ yb, const __bf16* __restrict__ wpT,
               float* __restrict__ out) {
    const int lane = threadIdx.x & 31;
    const int half = lane >> 4;
    const int l15  = lane & 15;
    const int row0 = blockIdx.x * 32;           // 256 M tiles of 32 rows
    const int nb   = blockIdx.y;                // 12 blocks of 64 cols

    f32x8 c[2][4];
    for (int m = 0; m < 2; ++m)
        for (int j = 0; j < 4; ++j)
            for (int r = 0; r < 8; ++r) c[m][j][r] = 0.0f;

    const __bf16* arow0 = yb + (size_t)(row0 + l15) * N_EMBD;
    const __bf16* arow1 = arow0 + (size_t)16 * N_EMBD;
    for (int k0 = 0; k0 < N_EMBD; k0 += 32) {
        bf16x16 a0 = make_afrag(arow0 + k0 + half * 16);
        bf16x16 a1 = make_afrag(arow1 + k0 + half * 16);
        for (int j = 0; j < 4; ++j) {
            int n = nb * 64 + j * 16 + l15;
            const __bf16* bp = wpT + (size_t)n * N_EMBD + k0 + half * 16;
            bf16x16 b = *(const bf16x16*)bp;
            c[0][j] = __builtin_amdgcn_wmma_f32_16x16x32_bf16(false, a0, false, b,
                                                              (short)0, c[0][j], false, false);
            c[1][j] = __builtin_amdgcn_wmma_f32_16x16x32_bf16(false, a1, false, b,
                                                              (short)0, c[1][j], false, false);
        }
    }
    for (int m = 0; m < 2; ++m) {
        for (int j = 0; j < 4; ++j) {
            int n = nb * 64 + j * 16 + l15;
            for (int r = 0; r < 8; ++r) {
                int row = row0 + m * 16 + r + half * 8;
                out[(size_t)row * N_EMBD + n] = c[m][j][r];
            }
        }
    }
}

// ---------------- launch ----------------

extern "C" void kernel_launch(void* const* d_in, const int* in_sizes, int n_in,
                              void* d_out, int out_size, void* d_ws, size_t ws_size,
                              hipStream_t stream) {
    const float* x      = (const float*)d_in[0];
    const float* w_attn = (const float*)d_in[1];
    const float* w_proj = (const float*)d_in[2];
    float* out = (float*)d_out;
    char* ws = (char*)d_ws;

    const size_t SZ_XB  = (size_t)BT * N_EMBD * 2;            // 12,582,912
    const size_t SZ_WAT = (size_t)C3 * N_EMBD * 2;            //  3,538,944
    const size_t SZ_WPT = (size_t)N_EMBD * N_EMBD * 2;        //  1,179,648
    const size_t SZ_HD  = (size_t)BATCH * N_HEAD * TSEQ * HEAD_DIM * 2; // 12,582,912

    size_t off = 0;
    __bf16* xb  = (__bf16*)(ws + off); off += SZ_XB;
    __bf16* waT = (__bf16*)(ws + off); off += SZ_WAT;
    __bf16* wpT = (__bf16*)(ws + off); off += SZ_WPT;
    __bf16* Q   = (__bf16*)(ws + off); off += SZ_HD;
    __bf16* Kk  = (__bf16*)(ws + off); off += SZ_HD;
    __bf16* Vt  = (__bf16*)(ws + off); off += SZ_HD;
    __bf16* yb  = (__bf16*)(ws + off); off += SZ_XB;

    {
        int n = BT * N_EMBD;
        cvt_f32_bf16<<<(n + 255) / 256, 256, 0, stream>>>(x, xb, n);
    }
    {
        int n = N_EMBD * C3;
        transpose_cvt<<<(n + 255) / 256, 256, 0, stream>>>(w_attn, waT, N_EMBD, C3);
    }
    {
        int n = N_EMBD * N_EMBD;
        transpose_cvt<<<(n + 255) / 256, 256, 0, stream>>>(w_proj, wpT, N_EMBD, N_EMBD);
    }
    qkv_gemm  <<<dim3(BT / 32, C3 / 64),          32, 0, stream>>>(xb, waT, Q, Kk, Vt);
    flash_attn<<<dim3(TSEQ / 32, BATCH * N_HEAD), 32, 0, stream>>>(Q, Kk, Vt, yb);
    proj_gemm <<<dim3(BT / 32, N_EMBD / 64),      32, 0, stream>>>(yb, wpT, out);
}
